// OnlineHardTripletLoss_34617436406313
// MI455X (gfx1250) — matrix operand order repaired and checked by reading, not compiled, not run on hardware
//
#include <hip/hip_runtime.h>
#include <hip/hip_bf16.h>
#include <stdint.h>

#define BDIM 8192
#define DDIM 128
#define NSPLIT 8
#define NCHUNK (BDIM / NSPLIT)      // 1024 columns per workgroup
#define NTILES (NCHUNK / 16)        // 64 B-tiles per chunk
#define MARGIN 0.3f
#define HN2_SENTINEL 1e18f          // (1e9)^2 sentinel in squared-distance domain

typedef __attribute__((ext_vector_type(16))) __bf16 v16bf;
typedef __attribute__((ext_vector_type(8)))  float  v8f;
typedef __attribute__((vector_size(16)))     int    v4i_vs;   // matches builtin param type

union Frag {
    v16bf v;
    uint4 q[2];
};

#ifndef __has_builtin
#define __has_builtin(x) 0
#endif

#if __has_builtin(__builtin_amdgcn_global_load_async_to_lds_b128)
#define HAVE_ASYNC_LDS 1
#else
#define HAVE_ASYNC_LDS 0
#endif

#if __has_builtin(__builtin_amdgcn_s_wait_asynccnt)
#define WAIT_ASYNC(n) __builtin_amdgcn_s_wait_asynccnt(n)
#else
#define WAIT_ASYNC(n) asm volatile("s_wait_asynccnt %0" ::"i"(n) : "memory")
#endif

// copy 16 bytes global -> LDS (async DMA when available)
__device__ __forceinline__ void copy16_to_lds(const void* g, void* l) {
#if HAVE_ASYNC_LDS
    __builtin_amdgcn_global_load_async_to_lds_b128((v4i_vs*)g, (v4i_vs*)l, 0, 0);
#else
    *(uint4*)l = *(const uint4*)g;
#endif
}

__device__ __forceinline__ uint16_t f32_to_bf16_rne(float x) {
    uint32_t u = __float_as_uint(x);
    uint32_t r = u + 0x7FFFu + ((u >> 16) & 1u);
    return (uint16_t)(r >> 16);
}

// ---------------------------------------------------------------------------
// Prep: split f32 -> (hi, lo) bf16 planes, per-row squared norm, init reducers
// ---------------------------------------------------------------------------
__global__ __launch_bounds__(DDIM) void triplet_prep(
    const float* __restrict__ emb,
    uint16_t* __restrict__ hi, uint16_t* __restrict__ lo,
    float* __restrict__ sq,
    float* __restrict__ hp2, float* __restrict__ hn2, int* __restrict__ pa)
{
    const int row = blockIdx.x;
    const int c   = threadIdx.x;            // 0..127
    const float x = emb[row * DDIM + c];

    uint16_t h  = f32_to_bf16_rne(x);
    float    hf = __uint_as_float((uint32_t)h << 16);
    uint16_t l  = f32_to_bf16_rne(x - hf);
    hi[row * DDIM + c] = h;
    lo[row * DDIM + c] = l;

    float s = x * x;
    #pragma unroll
    for (int m = 16; m >= 1; m >>= 1) s += __shfl_xor(s, m, 32);
    __shared__ float ls[4];
    if ((c & 31) == 0) ls[c >> 5] = s;
    __syncthreads();
    if (c == 0) {
        sq[row]  = ls[0] + ls[1] + ls[2] + ls[3];
        hp2[row] = 0.0f;          // masked-max sentinel (d^2 clamped >= 0)
        hn2[row] = HN2_SENTINEL;  // masked-min sentinel
        pa[row]  = 0;             // any-positive flag
    }
}

// ---------------------------------------------------------------------------
// Stage one 16-column B tile (hi+lo planes, 8 KB) into an LDS buffer.
// 256 threads x 2 chunks of 16B. LDS layout: hi rows [0,4096), lo [4096,8192),
// row-major, 256 B row stride per plane.
// ---------------------------------------------------------------------------
__device__ __forceinline__ void stage_tile(
    const uint16_t* __restrict__ hi, const uint16_t* __restrict__ lo,
    int j0, char* buf, int tid)
{
    #pragma unroll
    for (int s = 0; s < 2; ++s) {
        const int c     = tid * 2 + s;       // 0..511
        const int plane = c >> 8;            // 0 = hi, 1 = lo
        const int cc    = c & 255;
        const int row   = cc >> 4;           // 0..15  (column j0+row of X)
        const int col   = (cc & 15) * 8;     // element offset in the row
        const uint16_t* src = (plane ? lo : hi) + (size_t)(j0 + row) * DDIM + col;
        copy16_to_lds(src, buf + c * 16);
    }
}

// ---------------------------------------------------------------------------
// Main: split-bf16 WMMA Gram tiles + fused hardest-pos / hardest-neg mining.
// WG = 8 waves = 128 anchor rows, all sharing one 1024-column chunk whose
// B tiles are staged cooperatively in LDS (double-buffered async DMA).
// ---------------------------------------------------------------------------
__global__ __launch_bounds__(256) void triplet_main(
    const uint16_t* __restrict__ hi, const uint16_t* __restrict__ lo,
    const float* __restrict__ sq, const int* __restrict__ labels,
    float* __restrict__ hp2, float* __restrict__ hn2, int* __restrict__ pa)
{
    __shared__ __align__(16) char smem[2][8192];

    const int tid  = threadIdx.x;
    const int wave = tid >> 5;
    const int lane = tid & 31;
    const int half = lane >> 4;            // K sub-window / M+8 selector
    const int l16  = lane & 15;

    const int rowgrp  = blockIdx.x >> 3;               // 0..63
    const int chunk   = blockIdx.x & (NSPLIT - 1);     // 0..7
    const int i0      = rowgrp * 128 + wave * 16;      // this wave's 16 rows
    const int j_begin = chunk * NCHUNK;

    // --- A fragments (16 rows x K=128, hi & lo planes), register-resident ---
    // ISA A layout (16-bit, 16x32): lane = M row (mod 16); half 0 -> K {0..7,16..23},
    // half 1 -> K {8..15,24..31} within each K window.
    Frag ahi[4], alo[4];
    {
        const uint16_t* ph = hi + (size_t)(i0 + l16) * DDIM;
        const uint16_t* pl = lo + (size_t)(i0 + l16) * DDIM;
        #pragma unroll
        for (int kt = 0; kt < 4; ++kt) {
            const int c1 = kt * 32 + half * 8;
            ahi[kt].q[0] = *(const uint4*)(ph + c1);
            ahi[kt].q[1] = *(const uint4*)(ph + c1 + 16);
            alo[kt].q[0] = *(const uint4*)(pl + c1);
            alo[kt].q[1] = *(const uint4*)(pl + c1 + 16);
        }
    }

    // per-lane anchor metadata: this lane's 8 C-tile rows are M = r + 8*half
    float sqi[8];
    int   labi[8];
    #pragma unroll
    for (int r = 0; r < 8; ++r) {
        const int i = i0 + r + 8 * half;
        sqi[r]  = sq[i];
        labi[r] = labels[i];
    }

    float vhp[8], vhn[8];
    int vpa = 0;
    #pragma unroll
    for (int r = 0; r < 8; ++r) { vhp[r] = 0.0f; vhn[r] = HN2_SENTINEL; }

    stage_tile(hi, lo, j_begin, smem[0], tid);

    for (int t = 0; t < NTILES; ++t) {
        const int j0 = j_begin + t * 16;

        // prefetch next tile into the other buffer, then wait for this tile
        if (t + 1 < NTILES) {
            stage_tile(hi, lo, j0 + 16, smem[(t + 1) & 1], tid);
#if HAVE_ASYNC_LDS
            WAIT_ASYNC(2);   // leave only the 2 newest (next-tile) DMAs in flight
#endif
        } else {
#if HAVE_ASYNC_LDS
            WAIT_ASYNC(0);
#endif
        }
        __syncthreads();     // everyone's portion of tile t is visible

        const char* bh = smem[t & 1];
        const char* bl = bh + 4096;

        v8f acc = {};
        #pragma unroll
        for (int kt = 0; kt < 4; ++kt) {
            Frag bhi, blo;
            const int cb = (kt * 32 + half * 8) * 2;    // byte offset in a row
            bhi.q[0] = *(const uint4*)(bh + l16 * 256 + cb);
            bhi.q[1] = *(const uint4*)(bh + l16 * 256 + cb + 32);
            blo.q[0] = *(const uint4*)(bl + l16 * 256 + cb);
            blo.q[1] = *(const uint4*)(bl + l16 * 256 + cb + 32);
            // dot_f32 ~= hi*hi + hi*lo + lo*hi  (lo*lo ~ 2^-18, negligible)
            acc = __builtin_amdgcn_wmma_f32_16x16x32_bf16(false, ahi[kt].v, false, bhi.v,
                                                          (short)0, acc, false, false);
            acc = __builtin_amdgcn_wmma_f32_16x16x32_bf16(false, ahi[kt].v, false, blo.v,
                                                          (short)0, acc, false, false);
            acc = __builtin_amdgcn_wmma_f32_16x16x32_bf16(false, alo[kt].v, false, bhi.v,
                                                          (short)0, acc, false, false);
        }

        const int   jcol = j0 + l16;
        const int   labj = labels[jcol];
        const float sqj  = sq[jcol];

        // C layout: VGPR r, lanes 0-15 -> M=r, lanes 16-31 -> M=8+r; N = lane&15
        // Work in squared distances; sqrt deferred (monotone transform).
        #pragma unroll
        for (int r = 0; r < 8; ++r) {
            float d2 = fmaxf(sqi[r] + sqj - 2.0f * acc[r], 0.0f);
            const int  irow  = i0 + r + 8 * half;
            const bool same  = (labi[r] == labj);
            const bool ispos = same && (jcol != irow);
            if (ispos)  { vhp[r] = fmaxf(vhp[r], d2); vpa |= (1 << r); }
            if (!same)  { vhn[r] = fminf(vhn[r], d2); }
        }

        __syncthreads();     // all waves done with buffer (t&1) before reuse
    }

    // reduce across the 16 N-lanes of each half-wave (butterfly, masks < 16)
    #pragma unroll
    for (int r = 0; r < 8; ++r) {
        #pragma unroll
        for (int m = 1; m < 16; m <<= 1) {
            vhp[r] = fmaxf(vhp[r], __shfl_xor(vhp[r], m, 32));
            vhn[r] = fminf(vhn[r], __shfl_xor(vhn[r], m, 32));
        }
    }
    #pragma unroll
    for (int m = 1; m < 16; m <<= 1) vpa |= __shfl_xor(vpa, m, 32);

    // combine across N-chunks: float-as-int atomics (all values >= 0 -> exact order)
    if (l16 == 0) {
        #pragma unroll
        for (int r = 0; r < 8; ++r) {
            const int i = i0 + r + 8 * half;
            atomicMax((int*)(hp2 + i), __float_as_int(vhp[r]));
            atomicMin((int*)(hn2 + i), __float_as_int(vhn[r]));
            if (vpa & (1 << r)) atomicOr(pa + i, 1);
        }
    }
}

// ---------------------------------------------------------------------------
// Finalize: sqrt of squared extrema, hinge, valid-masked mean -> scalar loss
// ---------------------------------------------------------------------------
__global__ __launch_bounds__(256) void triplet_finalize(
    const float* __restrict__ hp2, const float* __restrict__ hn2,
    const int* __restrict__ pa, float* __restrict__ out)
{
    float total = 0.0f, cnt = 0.0f;
    for (int i = threadIdx.x; i < BDIM; i += 256) {
        const bool valid = (pa[i] != 0) && (hn2[i] < 1e16f);
        float per = sqrtf(hp2[i]) - sqrtf(hn2[i]) + MARGIN;
        per = per > 0.0f ? per : 0.0f;
        if (valid) { total += per; cnt += 1.0f; }
    }
    #pragma unroll
    for (int m = 16; m >= 1; m >>= 1) {
        total += __shfl_xor(total, m, 32);
        cnt   += __shfl_xor(cnt,   m, 32);
    }
    __shared__ float st[8], sc[8];
    const int w = threadIdx.x >> 5;
    if ((threadIdx.x & 31) == 0) { st[w] = total; sc[w] = cnt; }
    __syncthreads();
    if (threadIdx.x == 0) {
        float T = 0.0f, C = 0.0f;
        #pragma unroll
        for (int k = 0; k < 8; ++k) { T += st[k]; C += sc[k]; }
        out[0] = C > 0.0f ? T / C : 0.0f;
    }
}

// ---------------------------------------------------------------------------
extern "C" void kernel_launch(void* const* d_in, const int* in_sizes, int n_in,
                              void* d_out, int out_size, void* d_ws, size_t ws_size,
                              hipStream_t stream) {
    const float* emb    = (const float*)d_in[0];   // [8192,128] f32
    const int*   labels = (const int*)d_in[1];     // [8192] i32
    float*       out    = (float*)d_out;           // scalar f32

    char* w = (char*)d_ws;
    const size_t HL = (size_t)BDIM * DDIM * sizeof(uint16_t);  // 2 MB per plane
    uint16_t* hi  = (uint16_t*)(w);
    uint16_t* lo  = (uint16_t*)(w + HL);
    float*    sq  = (float*)(w + 2 * HL);
    float*    hp2 = (float*)(w + 2 * HL + (size_t)BDIM * 4);
    float*    hn2 = (float*)(w + 2 * HL + (size_t)BDIM * 8);
    int*      pa  = (int*)  (w + 2 * HL + (size_t)BDIM * 12);

    triplet_prep<<<BDIM, DDIM, 0, stream>>>(emb, hi, lo, sq, hp2, hn2, pa);
    triplet_main<<<(BDIM / 128) * NSPLIT, 256, 0, stream>>>(hi, lo, sq, labels, hp2, hn2, pa);
    triplet_finalize<<<1, 256, 0, stream>>>(hp2, hn2, pa, out);
}
